// MultiCatKernelSVM_23003844837706
// MI455X (gfx1250) — matrix operand order repaired and checked by reading, not compiled, not run on hardware
//
#include <hip/hip_runtime.h>
#include <hip/hip_bf16.h>

// MI455X plan:
//   out = exp(-gamma * (||xi||^2 + ||xj||^2 - 2 x@x^T)) @ W^T + b
// Matrix-unit bound (~68 GFLOP vs ~50MB HBM; K stays in 192MB L2).
// f16 hi/lo split GEMMs on v_wmma_f32_16x16x32_f16 (hi*hi + hi*lo + lo*hi).
// Operands pre-split to f16 planes once; GEMM tiles staged into LDS with
// gfx1250 global_load_async_to_lds_b128 (SADDR form: uniform SGPR base +
// loop-invariant 32-bit lane offset), double-buffered so the async DMA of
// tile k+1 overlaps the 24 WMMAs of tile k.

typedef _Float16 v16h __attribute__((ext_vector_type(16)));
typedef _Float16 v8h  __attribute__((ext_vector_type(8)));
typedef _Float16 v4h  __attribute__((ext_vector_type(4)));
typedef float    v8f  __attribute__((ext_vector_type(8)));
typedef float    v4f  __attribute__((ext_vector_type(4)));

#define N_SAMP 4096
#define D_IN   1024
#define N_OUT  1000
#define W_PAD  1024    // W rows padded to 1024 (rows >= 1000 zero-filled)
#define LDT    40      // LDS row stride in halves (80B, conflict-free reads)
#define TILE_BYTES (128 * LDT * 2)   // 10240B per 128x32 f16 tile

// ---------------- async copy (gfx1250) ----------------

__device__ __forceinline__ unsigned lds_off32(const void* p) {
    // Generic LDS address = {SHARED_BASE, offset}; low 32 bits are the LDS byte offset.
    return (unsigned)(unsigned long long)p;
}

// SADDR form: mem_addr = SGPR base + 32-bit lane byte offset.
__device__ __forceinline__ void async_b128(unsigned lds_addr, unsigned voff, const _Float16* sbase) {
    asm volatile("global_load_async_to_lds_b128 %0, %1, %2"
                 :: "v"(lds_addr), "v"(voff), "s"(sbase) : "memory");
}

__device__ __forceinline__ void wait_async0() {
    asm volatile("s_wait_asynccnt 0x0" ::: "memory");
}

// ---------------- fragment helpers ----------------

// A fragment, 16x32 f16: lanes 0-15 row M=lane hold K 0-7 & 16-23,
// lanes 16-31 row M=lane-16 hold K 8-15 & 24-31.
__device__ __forceinline__ v16h load_fragA(const _Float16* base, int rowBase, int lane) {
    int r  = rowBase + (lane & 15);
    int kh = (lane >> 4) * 8;
    v8h c0 = *reinterpret_cast<const v8h*>(base + r * LDT + kh);
    v8h c1 = *reinterpret_cast<const v8h*>(base + r * LDT + kh + 16);
    return __builtin_shufflevector(c0, c1, 0,1,2,3,4,5,6,7,8,9,10,11,12,13,14,15);
}

// B fragment, 32x16 f16: lanes 0-15 col N=lane hold K 0-7 & 16-23,
// lanes 16-31 col N=lane-16 hold K 8-15 & 24-31.
__device__ __forceinline__ v16h load_fragB(const _Float16* base, int rowBase, int lane) {
    int r  = rowBase + (lane & 15);
    int kh = (lane >> 4) * 8;
    v8h c0 = *reinterpret_cast<const v8h*>(base + r * LDT + kh);
    v8h c1 = *reinterpret_cast<const v8h*>(base + r * LDT + kh + 16);
    return __builtin_shufflevector(c0, c1, 0,1,2,3,4,5,6,7,8,9,10,11,12,13,14,15);
}

// ---------------- prep kernels ----------------

__global__ __launch_bounds__(256) void svm_rownorm_kernel(const float* __restrict__ x,
                                                          float* __restrict__ sq) {
    int row  = blockIdx.x * 8 + (threadIdx.x >> 5);
    int lane = threadIdx.x & 31;
    const float* xr = x + (size_t)row * D_IN;
    float s = 0.0f;
#pragma unroll 4
    for (int i = lane; i < D_IN; i += 32) {
        float v = xr[i];
        s = fmaf(v, v, s);
    }
#pragma unroll
    for (int off = 16; off > 0; off >>= 1) s += __shfl_xor(s, off, 32);
    if (lane == 0) sq[row] = s;
}

__global__ __launch_bounds__(256) void svm_split_x_kernel(const float* __restrict__ x,
                                                          _Float16* __restrict__ xhi,
                                                          _Float16* __restrict__ xlo) {
    size_t i4 = (size_t)blockIdx.x * 256 + threadIdx.x;     // float4 chunk id
    v4f v = *(reinterpret_cast<const v4f*>(x) + i4);
    v4h h, l;
#pragma unroll
    for (int j = 0; j < 4; ++j) {
        _Float16 hj = (_Float16)v[j];
        _Float16 lj = (_Float16)(v[j] - (float)hj);
        h[j] = hj;
        l[j] = lj;
    }
    *(reinterpret_cast<v4h*>(xhi) + i4) = h;
    *(reinterpret_cast<v4h*>(xlo) + i4) = l;
}

__global__ __launch_bounds__(256) void svm_split_w_kernel(const float* __restrict__ W,
                                                          _Float16* __restrict__ whi,
                                                          _Float16* __restrict__ wlo) {
    size_t i4  = (size_t)blockIdx.x * 256 + threadIdx.x;    // over W_PAD*4096/4 chunks
    int   row  = (int)(i4 >> 10);                           // 1024 float4 chunks per row
    v4f v = {0.0f, 0.0f, 0.0f, 0.0f};
    if (row < N_OUT) v = *(reinterpret_cast<const v4f*>(W) + i4);
    v4h h, l;
#pragma unroll
    for (int j = 0; j < 4; ++j) {
        _Float16 hj = (_Float16)v[j];
        _Float16 lj = (_Float16)(v[j] - (float)hj);
        h[j] = hj;
        l[j] = lj;
    }
    *(reinterpret_cast<v4h*>(whi) + i4) = h;
    *(reinterpret_cast<v4h*>(wlo) + i4) = l;
}

// ---------------- GEMM1: K = exp(-g*(sqm+sqn-2 x x^T)) -> hi/lo f16 ----------------

__global__ __launch_bounds__(256) void svm_gram_exp_kernel(const _Float16* __restrict__ xhi,
                                                           const _Float16* __restrict__ xlo,
                                                           const float* __restrict__ gamma_p,
                                                           const float* __restrict__ sq,
                                                           _Float16* __restrict__ Khi,
                                                           _Float16* __restrict__ Klo) {
    __shared__ _Float16 smem[2][4][128 * LDT];   // [buf][Ahi,Alo,Bhi,Blo]

    const int tid  = threadIdx.x;
    const int lane = tid & 31;
    const int wid  = tid >> 5;
    const int mw   = wid >> 1;
    const int nw   = wid & 1;
    const int bm   = (int)blockIdx.y;
    const int bn   = (int)blockIdx.x;

    // per-thread async-copy slots: 512 16B chunks per tile, 2 per thread
    unsigned lbase = lds_off32(&smem[0][0][0]);
    unsigned ldsoff[2], offA[2], offB[2];     // byte offsets (loop-invariant)
#pragma unroll
    for (int i = 0; i < 2; ++i) {
        int c = tid + 256 * i;            // 0..511
        int row = c >> 2, ch = c & 3;     // 16B chunk within 64B row
        ldsoff[i] = (unsigned)(row * (LDT * 2) + ch * 16);
        offA[i]   = (unsigned)(((bm * 128 + row) * D_IN + ch * 8) * 2);
        offB[i]   = (unsigned)(((bn * 128 + row) * D_IN + ch * 8) * 2);
    }

    auto issue = [&](int buf, int k0) {
        unsigned b = lbase + (unsigned)buf * 4 * TILE_BYTES;
        const _Float16* bh = xhi + k0;    // uniform SGPR bases (k0 folded via SALU)
        const _Float16* bl = xlo + k0;
#pragma unroll
        for (int i = 0; i < 2; ++i) {
            unsigned l = b + ldsoff[i];
            async_b128(l + 0 * TILE_BYTES, offA[i], bh);
            async_b128(l + 1 * TILE_BYTES, offA[i], bl);
            async_b128(l + 2 * TILE_BYTES, offB[i], bh);
            async_b128(l + 3 * TILE_BYTES, offB[i], bl);
        }
    };

    v8f acc[2][4] = {};

    issue(0, 0);
    wait_async0();
    __syncthreads();

    for (int k0 = 0; k0 < D_IN; k0 += 32) {
        int cur = (k0 >> 5) & 1;
        if (k0 + 32 < D_IN) issue(cur ^ 1, k0 + 32);

        const _Float16* Ahi = smem[cur][0];
        const _Float16* Alo = smem[cur][1];
        const _Float16* Bhi = smem[cur][2];
        const _Float16* Blo = smem[cur][3];

        v16h bh[4], bl[4];
#pragma unroll
        for (int tj = 0; tj < 4; ++tj) {
            bh[tj] = load_fragB(Bhi, nw * 64 + tj * 16, lane);
            bl[tj] = load_fragB(Blo, nw * 64 + tj * 16, lane);
        }
#pragma unroll
        for (int ti = 0; ti < 2; ++ti) {
            v16h ah = load_fragA(Ahi, mw * 32 + ti * 16, lane);
            v16h al = load_fragA(Alo, mw * 32 + ti * 16, lane);
#pragma unroll
            for (int tj = 0; tj < 4; ++tj) {
                v8f c = acc[ti][tj];
                c = __builtin_amdgcn_wmma_f32_16x16x32_f16(false, ah, false, bh[tj], (short)0, c, false, false);
                c = __builtin_amdgcn_wmma_f32_16x16x32_f16(false, ah, false, bl[tj], (short)0, c, false, false);
                c = __builtin_amdgcn_wmma_f32_16x16x32_f16(false, al, false, bh[tj], (short)0, c, false, false);
                acc[ti][tj] = c;
            }
        }
        wait_async0();
        __syncthreads();
    }

    const float gamma = gamma_p[0];
    const int nl = lane & 15, half = lane >> 4;
#pragma unroll
    for (int ti = 0; ti < 2; ++ti) {
#pragma unroll
        for (int tj = 0; tj < 4; ++tj) {
            int n = bn * 128 + nw * 64 + tj * 16 + nl;
            float sqn = sq[n];
#pragma unroll
            for (int r = 0; r < 8; ++r) {
                int m = bm * 128 + mw * 32 + ti * 16 + half * 8 + r;
                float d2 = fmaxf(sq[m] + sqn - 2.0f * acc[ti][tj][r], 0.0f);
                float kv = __expf(-gamma * d2);
                _Float16 kh = (_Float16)kv;
                _Float16 kl = (_Float16)(kv - (float)kh);
                Khi[(size_t)m * N_SAMP + n] = kh;
                Klo[(size_t)m * N_SAMP + n] = kl;
            }
        }
    }
}

// ---------------- GEMM2: out = K @ W^T + b ----------------

__global__ __launch_bounds__(256) void svm_head_kernel(const _Float16* __restrict__ Khi,
                                                       const _Float16* __restrict__ Klo,
                                                       const _Float16* __restrict__ Whi,
                                                       const _Float16* __restrict__ Wlo,
                                                       const float* __restrict__ bias,
                                                       float* __restrict__ out) {
    __shared__ _Float16 smem[2][4][128 * LDT];

    const int tid  = threadIdx.x;
    const int lane = tid & 31;
    const int wid  = tid >> 5;
    const int mw   = wid >> 1;
    const int nw   = wid & 1;
    const int bm   = (int)blockIdx.y;
    const int bn   = (int)blockIdx.x;

    unsigned lbase = lds_off32(&smem[0][0][0]);
    unsigned ldsoff[2], offA[2], offB[2];
#pragma unroll
    for (int i = 0; i < 2; ++i) {
        int c = tid + 256 * i;
        int row = c >> 2, ch = c & 3;
        ldsoff[i] = (unsigned)(row * (LDT * 2) + ch * 16);
        offA[i]   = (unsigned)(((bm * 128 + row) * N_SAMP + ch * 8) * 2);
        offB[i]   = (unsigned)(((bn * 128 + row) * N_SAMP + ch * 8) * 2);  // W padded
    }

    auto issue = [&](int buf, int k0) {
        unsigned b = lbase + (unsigned)buf * 4 * TILE_BYTES;
        const _Float16* akh = Khi + k0;
        const _Float16* akl = Klo + k0;
        const _Float16* bwh = Whi + k0;
        const _Float16* bwl = Wlo + k0;
#pragma unroll
        for (int i = 0; i < 2; ++i) {
            unsigned l = b + ldsoff[i];
            async_b128(l + 0 * TILE_BYTES, offA[i], akh);
            async_b128(l + 1 * TILE_BYTES, offA[i], akl);
            async_b128(l + 2 * TILE_BYTES, offB[i], bwh);
            async_b128(l + 3 * TILE_BYTES, offB[i], bwl);
        }
    };

    v8f acc[2][4] = {};

    issue(0, 0);
    wait_async0();
    __syncthreads();

    for (int k0 = 0; k0 < N_SAMP; k0 += 32) {
        int cur = (k0 >> 5) & 1;
        if (k0 + 32 < N_SAMP) issue(cur ^ 1, k0 + 32);

        const _Float16* Ahi = smem[cur][0];
        const _Float16* Alo = smem[cur][1];
        const _Float16* Bhi = smem[cur][2];
        const _Float16* Blo = smem[cur][3];

        v16h bh[4], bl[4];
#pragma unroll
        for (int tj = 0; tj < 4; ++tj) {
            bh[tj] = load_fragB(Bhi, nw * 64 + tj * 16, lane);
            bl[tj] = load_fragB(Blo, nw * 64 + tj * 16, lane);
        }
#pragma unroll
        for (int ti = 0; ti < 2; ++ti) {
            v16h ah = load_fragA(Ahi, mw * 32 + ti * 16, lane);
            v16h al = load_fragA(Alo, mw * 32 + ti * 16, lane);
#pragma unroll
            for (int tj = 0; tj < 4; ++tj) {
                v8f c = acc[ti][tj];
                c = __builtin_amdgcn_wmma_f32_16x16x32_f16(false, ah, false, bh[tj], (short)0, c, false, false);
                c = __builtin_amdgcn_wmma_f32_16x16x32_f16(false, ah, false, bl[tj], (short)0, c, false, false);
                c = __builtin_amdgcn_wmma_f32_16x16x32_f16(false, al, false, bh[tj], (short)0, c, false, false);
                acc[ti][tj] = c;
            }
        }
        wait_async0();
        __syncthreads();
    }

    const int nl = lane & 15, half = lane >> 4;
#pragma unroll
    for (int ti = 0; ti < 2; ++ti) {
#pragma unroll
        for (int tj = 0; tj < 4; ++tj) {
            int n = bn * 128 + nw * 64 + tj * 16 + nl;
            if (n < N_OUT) {
                float bn_v = bias[n];
#pragma unroll
                for (int r = 0; r < 8; ++r) {
                    int m = bm * 128 + mw * 32 + ti * 16 + half * 8 + r;
                    out[(size_t)m * N_OUT + n] = acc[ti][tj][r] + bn_v;
                }
            }
        }
    }
}

// ---------------- launch ----------------

extern "C" void kernel_launch(void* const* d_in, const int* in_sizes, int n_in,
                              void* d_out, int out_size, void* d_ws, size_t ws_size,
                              hipStream_t stream) {
    const float* x     = (const float*)d_in[0];   // [4096,1024]
    const float* gamma = (const float*)d_in[1];   // [1]
    const float* W     = (const float*)d_in[2];   // [1000,4096]
    const float* b     = (const float*)d_in[3];   // [1000]
    float* out = (float*)d_out;                   // [4096,1000]

    char* ws = (char*)d_ws;
    float*    sq  = (float*)ws;                                   //  16KB
    _Float16* Xhi = (_Float16*)(ws + (64ull << 10));              //   8MB
    _Float16* Xlo = Xhi + (size_t)N_SAMP * D_IN;                  //   8MB
    _Float16* Whi = Xlo + (size_t)N_SAMP * D_IN;                  //   8MB (padded rows)
    _Float16* Wlo = Whi + (size_t)W_PAD * N_SAMP;                 //   8MB
    _Float16* Khi = Wlo + (size_t)W_PAD * N_SAMP;                 //  32MB
    _Float16* Klo = Khi + (size_t)N_SAMP * N_SAMP;                //  32MB

    svm_rownorm_kernel<<<N_SAMP / 8, 256, 0, stream>>>(x, sq);
    svm_split_x_kernel<<<(N_SAMP * D_IN / 4) / 256, 256, 0, stream>>>(x, Xhi, Xlo);
    svm_split_w_kernel<<<((size_t)W_PAD * N_SAMP / 4) / 256, 256, 0, stream>>>(W, Whi, Wlo);
    svm_gram_exp_kernel<<<dim3(N_SAMP / 128, N_SAMP / 128), 256, 0, stream>>>(Xhi, Xlo, gamma, sq, Khi, Klo);
    svm_head_kernel<<<dim3((N_OUT + 127) / 128, N_SAMP / 128), 256, 0, stream>>>(Khi, Klo, Whi, Wlo, b, out);
}
